// _MobileViTBlock_28321014350402
// MI455X (gfx1250) — matrix operand order repaired and checked by reading, not compiled, not run on hardware
//
#include <hip/hip_runtime.h>
#include <hip/hip_bf16.h>
#include <math.h>

// ---------------- CDNA5 WMMA types ----------------
typedef __bf16 bf16_t;
typedef __attribute__((ext_vector_type(16))) __bf16 v16bf;
typedef __attribute__((ext_vector_type(8)))  float  v8f;

union V16U { v16bf v; uint4 q[2]; };

__device__ inline v8f v8f_zero() {
  v8f z = {0.f,0.f,0.f,0.f,0.f,0.f,0.f,0.f};
  return z;
}

// A fragment (16x32 bf16, MxK) from row-major [M, lda] bf16.
// lane 0-15:  m=lane,    K = k0+0..15
// lane 16-31: m=lane-16, K = k0+16..31
__device__ inline v16bf wmma_load_a(const bf16_t* A, int lda, int m0, int k0, int lane) {
  int m  = m0 + (lane & 15);
  int ks = k0 + ((lane >> 4) << 4);
  const bf16_t* p = A + (size_t)m * lda + ks;
  V16U u;
  u.q[0] = *(const uint4*)(p);        // K = ks..ks+7
  u.q[1] = *(const uint4*)(p + 8);    // K = ks+8..ks+15
  return u.v;
}

// B fragment (32x16 bf16, KxN) from pre-packed layout:
// packed[((kt*ntN + nt)*32 + lane)*16 + i] = B[kt*32 + (lane>>4)*16 + i][nt*16 + (lane&15)]
__device__ inline v16bf wmma_load_b_packed(const bf16_t* Bp, int ntN, int kt, int nt, int lane) {
  const bf16_t* p = Bp + (((size_t)kt * ntN + nt) * 32 + lane) * 16;
  V16U u;
  u.q[0] = *(const uint4*)(p);
  u.q[1] = *(const uint4*)(p + 8);
  return u.v;
}

// B fragment where B[k][n] = R[n][k] with R row-major [N, ldr]: contiguous along k.
__device__ inline v16bf wmma_load_b_t(const bf16_t* R, int ldr, int n0, int kt, int lane) {
  int n  = n0 + (lane & 15);
  int kb = (kt << 5) + ((lane >> 4) << 4);
  const bf16_t* p = R + (size_t)n * ldr + kb;
  V16U u;
  u.q[0] = *(const uint4*)(p);
  u.q[1] = *(const uint4*)(p + 8);
  return u.v;
}

__device__ inline v8f wmma_bf16(v16bf a, v16bf b, v8f c) {
  return __builtin_amdgcn_wmma_f32_16x16x32_bf16(false, a, false, b, (short)0, c, false, false);
}

// ---------------- CDNA5 async global->LDS staging ----------------
// Copy a 16x32 bf16 A tile (1KB) into LDS: 32 lanes x 32B, two b128 asyncs per
// lane (INST_OFFSET applies to both the global and LDS sides per the ISA).
__device__ inline void stage_a_async(const bf16_t* A, int lda, int m0, int k0,
                                     bf16_t* lds_tile, int lane) {
  int row  = lane >> 1;
  int half = lane & 1;
  const bf16_t* g = A + (size_t)(m0 + row) * lda + k0 + half * 16;
  uint32_t lo = (uint32_t)(uintptr_t)(lds_tile + row * 32 + half * 16);
  uint64_t ga = (uint64_t)(uintptr_t)g;
  asm volatile(
      "global_load_async_to_lds_b128 %0, %1, off\n\t"
      "global_load_async_to_lds_b128 %0, %1, off offset:16"
      :: "v"(lo), "v"(ga) : "memory");
}

__device__ inline void wait_async0() {
  asm volatile("s_wait_asynccnt 0" ::: "memory");
}

// Shared async-staged, double-buffered 16x16-tile GEMM K-loop.
// All 4 waves of the block share the same M tile (mt = blockIdx.x); wave 0
// async-copies the A tile for step kt+1 into LDS while everyone computes kt.
__device__ inline v8f gemm_tile_async(const bf16_t* A, int lda, const bf16_t* Bp,
                                      int ktiles, int ntN, int mt, int nt,
                                      int lane, int wv) {
  __shared__ __align__(16) bf16_t a_s[2][16][32];
  if (wv == 0) {
    stage_a_async(A, lda, mt << 4, 0, &a_s[0][0][0], lane);
    wait_async0();
  }
  __syncthreads();
  v8f acc = v8f_zero();
  for (int kt = 0; kt < ktiles; ++kt) {
    int buf = kt & 1;
    if (wv == 0 && kt + 1 < ktiles)
      stage_a_async(A, lda, mt << 4, (kt + 1) << 5, &a_s[buf ^ 1][0][0], lane);
    v16bf a = wmma_load_a(&a_s[buf][0][0], 32, 0, 0, lane);
    acc = wmma_bf16(a, wmma_load_b_packed(Bp, ntN, kt, nt, lane), acc);
    if (wv == 0 && kt + 1 < ktiles) wait_async0();
    __syncthreads();
  }
  return acc;
}

// ---------------- problem constants ----------------
// B=128 C=256 H=W=32 TD=192 NHEADS=4 hd=48 FF=384 PH=PW=4 Nh=Nw=8 Np=64 M=8192

// ---------------- 1) patch-mean pool: x -> xm bf16 [8192, 256] ----------------
__global__ void pool_kernel(const float* __restrict__ x, bf16_t* __restrict__ xm) {
  int t = blockIdx.x * blockDim.x + threadIdx.x;       // 2M threads
  int p = t & 63;
  int c = (t >> 6) & 255;
  int b = t >> 14;
  int h0 = (p >> 3) << 2;
  int w0 = (p & 7) << 2;
  const float* xp = x + (((size_t)b * 256 + c) << 10) + h0 * 32 + w0;
  float s = 0.f;
#pragma unroll
  for (int i = 0; i < 4; ++i) {
    float4 v = *(const float4*)(xp + i * 32);
    s += v.x + v.y + v.z + v.w;
  }
  xm[(((size_t)b << 6) + p) * 256 + c] = (bf16_t)(s * 0.0625f);
}

// ---------------- 2) weight packing into B-fragment layout ----------------
__global__ void pack_b_kernel(const float* __restrict__ src, bf16_t* __restrict__ dst,
                              int K, int N, int trans, int ntN) {
  int lane = threadIdx.x;
  int kt = blockIdx.x / ntN;
  int nt = blockIdx.x % ntN;
  int nn = nt * 16 + (lane & 15);
  int kb = (kt << 5) + ((lane >> 4) << 4);
  bf16_t* out = dst + ((size_t)blockIdx.x * 32 + lane) * 16;
#pragma unroll
  for (int i = 0; i < 16; ++i) {
    int kk = kb + i;
    float v = 0.f;
    if (kk < K && nn < N)
      v = trans ? src[(size_t)nn * K + kk] : src[(size_t)kk * N + nn];
    out[i] = (bf16_t)v;
  }
}

// ---------------- 3) GEMM (N=192) + optional bias/residual + fused LayerNorm ----------------
__global__ __launch_bounds__(128) void gemm_ln_kernel(
    const bf16_t* __restrict__ A, int lda, const bf16_t* __restrict__ Bp, int ktiles,
    const float* __restrict__ bias, const float* __restrict__ resid,
    const float* __restrict__ alphap, float* __restrict__ outF,
    const float* __restrict__ lng, const float* __restrict__ lnb,
    bf16_t* __restrict__ outB) {
  __shared__ float sm[16][192];
  __shared__ float mu[16], rs[16];
  int lane = threadIdx.x & 31;
  int wv   = threadIdx.x >> 5;
  int m0   = blockIdx.x << 4;

  v8f acc[3] = {v8f_zero(), v8f_zero(), v8f_zero()};
  for (int kt = 0; kt < ktiles; ++kt) {
    v16bf a = wmma_load_a(A, lda, m0, kt << 5, lane);
#pragma unroll
    for (int j = 0; j < 3; ++j)
      acc[j] = wmma_bf16(a, wmma_load_b_packed(Bp, 12, kt, wv + 4 * j, lane), acc[j]);
  }

  float alpha = alphap ? alphap[0] : 0.f;
  int cn = lane & 15;
  int mo = (lane >> 4) << 3;
#pragma unroll
  for (int j = 0; j < 3; ++j) {
    int nt = wv + 4 * j;
    int col = (nt << 4) + cn;
    float bv = bias ? bias[col] : 0.f;
#pragma unroll
    for (int r = 0; r < 8; ++r) {
      int row = mo + r;
      float v = acc[j][r] + bv;
      if (resid) v = resid[(size_t)(m0 + row) * 192 + col] + alpha * v;
      sm[row][col] = v;
      outF[(size_t)(m0 + row) * 192 + col] = v;
    }
  }
  __syncthreads();

  if (threadIdx.x < 16) {
    int row = threadIdx.x;
    float s = 0.f;
    for (int c2 = 0; c2 < 192; ++c2) s += sm[row][c2];
    float m = s * (1.f / 192.f);
    float v2 = 0.f;
    for (int c2 = 0; c2 < 192; ++c2) { float d = sm[row][c2] - m; v2 += d * d; }
    mu[row] = m;
    rs[row] = rsqrtf(v2 * (1.f / 192.f) + 1e-5f);
  }
  __syncthreads();

  for (int idx = threadIdx.x; idx < 16 * 192; idx += 128) {
    int row = idx / 192, col = idx % 192;
    float v = (sm[row][col] - mu[row]) * rs[row] * lng[col] + lnb[col];
    outB[(size_t)(m0 + row) * 192 + col] = (bf16_t)v;
  }
}

// ---------------- 4) per-tile GEMMs using the shared async-staged K-loop ----------------
__global__ __launch_bounds__(128) void gemm_bias_f32_kernel(
    const bf16_t* __restrict__ A, int lda, const bf16_t* __restrict__ Bp, int ktiles, int ntN,
    const float* __restrict__ bias, float* __restrict__ C, int ldc) {
  int lane = threadIdx.x & 31, wv = threadIdx.x >> 5;
  int mt = blockIdx.x, nt = blockIdx.y * 4 + wv;
  v8f acc = gemm_tile_async(A, lda, Bp, ktiles, ntN, mt, nt, lane, wv);
  int cn = lane & 15, mo = (lane >> 4) << 3;
  int col = (nt << 4) + cn;
  float bv = bias ? bias[col] : 0.f;
#pragma unroll
  for (int r = 0; r < 8; ++r)
    C[(size_t)((mt << 4) + mo + r) * ldc + col] = acc[r] + bv;
}

__global__ __launch_bounds__(128) void gemm_bias_bf16_kernel(
    const bf16_t* __restrict__ A, int lda, const bf16_t* __restrict__ Bp, int ktiles, int ntN,
    const float* __restrict__ bias, bf16_t* __restrict__ C, int ldc) {
  int lane = threadIdx.x & 31, wv = threadIdx.x >> 5;
  int mt = blockIdx.x, nt = blockIdx.y * 4 + wv;
  v8f acc = gemm_tile_async(A, lda, Bp, ktiles, ntN, mt, nt, lane, wv);
  int cn = lane & 15, mo = (lane >> 4) << 3;
  int col = (nt << 4) + cn;
  float bv = bias[col];
#pragma unroll
  for (int r = 0; r < 8; ++r)
    C[(size_t)((mt << 4) + mo + r) * ldc + col] = (bf16_t)(acc[r] + bv);
}

__global__ __launch_bounds__(128) void gemm_gelu_bf16_kernel(
    const bf16_t* __restrict__ A, int lda, const bf16_t* __restrict__ Bp, int ktiles, int ntN,
    const float* __restrict__ bias, bf16_t* __restrict__ C, int ldc) {
  int lane = threadIdx.x & 31, wv = threadIdx.x >> 5;
  int mt = blockIdx.x, nt = blockIdx.y * 4 + wv;
  v8f acc = gemm_tile_async(A, lda, Bp, ktiles, ntN, mt, nt, lane, wv);
  int cn = lane & 15, mo = (lane >> 4) << 3;
  int col = (nt << 4) + cn;
  float bv = bias[col];
#pragma unroll
  for (int r = 0; r < 8; ++r) {
    float xg = acc[r] + bv;
    float g  = 0.5f * xg * (1.f + erff(xg * 0.70710678118654752f));   // exact GeLU
    C[(size_t)((mt << 4) + mo + r) * ldc + col] = (bf16_t)g;
  }
}

__global__ __launch_bounds__(128) void gemm_resid_bf16_kernel(
    const bf16_t* __restrict__ A, int lda, const bf16_t* __restrict__ Bp, int ktiles, int ntN,
    const float* __restrict__ bias, const float* __restrict__ resid,
    bf16_t* __restrict__ C) {   // N = 192 fixed
  int lane = threadIdx.x & 31, wv = threadIdx.x >> 5;
  int mt = blockIdx.x, nt = blockIdx.y * 4 + wv;
  v8f acc = gemm_tile_async(A, lda, Bp, ktiles, ntN, mt, nt, lane, wv);
  int cn = lane & 15, mo = (lane >> 4) << 3;
  int col = (nt << 4) + cn;
  float bv = bias[col];
#pragma unroll
  for (int r = 0; r < 8; ++r) {
    size_t idx = (size_t)((mt << 4) + mo + r) * 192 + col;
    C[idx] = (bf16_t)(resid[idx] + acc[r] + bv);
  }
}

// ---------------- 5) attention: one block per (batch, head) ----------------
__global__ __launch_bounds__(128) void attn_kernel(
    const bf16_t* __restrict__ qkv,  // [8192, 576] bf16 = [B*64][3,4,48]
    const float* __restrict__ rpb,   // [225, 4]
    bf16_t* __restrict__ oB) {       // [8192, 192] bf16
  __shared__ __align__(16) bf16_t q_s[64][64];
  __shared__ __align__(16) bf16_t k_s[64][64];
  __shared__ __align__(16) bf16_t vT_s[64][64];
  __shared__ float sc[64][64];
  __shared__ __align__(16) bf16_t p_s[64][64];

  int b = blockIdx.x >> 2, h = blockIdx.x & 3;
  int tid = threadIdx.x, lane = tid & 31, wv = tid >> 5;
  const bf16_t* base = qkv + ((size_t)b << 6) * 576 + h * 48;

  for (int idx = tid; idx < 64 * 64; idx += 128) {
    int tok = idx >> 6, d = idx & 63;
    bf16_t qv = (bf16_t)0.f, kv = (bf16_t)0.f, vv = (bf16_t)0.f;
    if (d < 48) {
      const bf16_t* pp = base + (size_t)tok * 576 + d;
      qv = pp[0]; kv = pp[192]; vv = pp[384];
    }
    q_s[tok][d]  = qv;
    k_s[tok][d]  = kv;
    vT_s[d][tok] = vv;
  }
  __syncthreads();

  // scores = q k^T * scale + rpb   (K dim = 48 padded to 64)
  const float scale = 0.14433756729740643f;  // 1/sqrt(48)
  int mt = wv;
  v8f accs[4] = {v8f_zero(), v8f_zero(), v8f_zero(), v8f_zero()};
  for (int kt = 0; kt < 2; ++kt) {
    v16bf a = wmma_load_a(&q_s[0][0], 64, mt << 4, kt << 5, lane);
#pragma unroll
    for (int nt = 0; nt < 4; ++nt)
      accs[nt] = wmma_bf16(a, wmma_load_b_t(&k_s[0][0], 64, nt << 4, kt, lane), accs[nt]);
  }
  int cn = lane & 15, mo = (lane >> 4) << 3;
#pragma unroll
  for (int nt = 0; nt < 4; ++nt) {
#pragma unroll
    for (int r = 0; r < 8; ++r) {
      int m = (mt << 4) + mo + r;
      int n = (nt << 4) + cn;
      int dh = (m >> 3) - (n >> 3) + 7;
      int dw = (m & 7) - (n & 7) + 7;
      sc[m][n] = accs[nt][r] * scale + rpb[(dh * 15 + dw) * 4 + h];
    }
  }
  __syncthreads();

  if (tid < 64) {
    float mx = -1e30f;
    for (int n = 0; n < 64; ++n) mx = fmaxf(mx, sc[tid][n]);
    float s = 0.f;
    for (int n = 0; n < 64; ++n) { float e = expf(sc[tid][n] - mx); sc[tid][n] = e; s += e; }
    float inv = 1.f / s;
    for (int n = 0; n < 64; ++n) p_s[tid][n] = (bf16_t)(sc[tid][n] * inv);
  }
  __syncthreads();

  // o = P @ V  (K = 64 tokens, N = 48)
  v8f acco[3] = {v8f_zero(), v8f_zero(), v8f_zero()};
  for (int kt = 0; kt < 2; ++kt) {
    v16bf a = wmma_load_a(&p_s[0][0], 64, mt << 4, kt << 5, lane);
#pragma unroll
    for (int j = 0; j < 3; ++j)
      acco[j] = wmma_bf16(a, wmma_load_b_t(&vT_s[0][0], 64, j << 4, kt, lane), acco[j]);
  }
#pragma unroll
  for (int j = 0; j < 3; ++j) {
#pragma unroll
    for (int r = 0; r < 8; ++r) {
      int m = (mt << 4) + mo + r;
      int col = h * 48 + (j << 4) + cn;
      oB[(((size_t)b << 6) + m) * 192 + col] = (bf16_t)(acco[j][r]);
    }
  }
}

// ---------------- 6) final broadcast add: out = x + y[b,p,c] ----------------
__global__ void final_add_kernel(const float* __restrict__ x, const float* __restrict__ y,
                                 float* __restrict__ out) {
  int t = blockIdx.x * blockDim.x + threadIdx.x;  // 8.4M quads
  int w4 = t & 7;
  int h  = (t >> 3) & 31;
  int c  = (t >> 8) & 255;
  int b  = t >> 16;
  int p  = ((h >> 2) << 3) + w4;
  size_t off = ((((size_t)b << 8) + c) << 10) + (h << 5) + (w4 << 2);
  float add = y[((((size_t)b << 6) + p) << 8) + c];
  float4 xv = *(const float4*)(x + off);
  xv.x += add; xv.y += add; xv.z += add; xv.w += add;
  *(float4*)(out + off) = xv;
}

// ---------------- launch ----------------
extern "C" void kernel_launch(void* const* d_in, const int* in_sizes, int n_in,
                              void* d_out, int out_size, void* d_ws, size_t ws_size,
                              hipStream_t stream) {
  const float* x       = (const float*)d_in[0];
  const float* conv1_w = (const float*)d_in[1];   // [192, 256]
  const float* conv2_w = (const float*)d_in[2];   // [256, 192]
  const float* qkv_w   = (const float*)d_in[3];   // [192, 576]
  const float* qkv_b   = (const float*)d_in[4];
  const float* proj_w  = (const float*)d_in[5];   // [192, 192]
  const float* proj_b  = (const float*)d_in[6];
  const float* ffn_w1  = (const float*)d_in[7];   // [192, 384]
  const float* ffn_b1  = (const float*)d_in[8];
  const float* ffn_w2  = (const float*)d_in[9];   // [384, 192]
  const float* ffn_b2  = (const float*)d_in[10];
  const float* ln1_g   = (const float*)d_in[11];
  const float* ln1_b   = (const float*)d_in[12];
  const float* ln2_g   = (const float*)d_in[13];
  const float* ln2_b   = (const float*)d_in[14];
  const float* alpha   = (const float*)d_in[15];
  const float* rpb     = (const float*)d_in[16];
  float* outp = (float*)d_out;

  char* ws = (char*)d_ws;
  size_t off = 0;
  auto take = [&](size_t bytes) { size_t r = off; off = (off + bytes + 255) & ~(size_t)255; return r; };
  const size_t M = 8192;
  bf16_t* xm     = (bf16_t*)(ws + take(M * 256 * 2));
  float*  tokens = (float* )(ws + take(M * 192 * 4));
  bf16_t* tbf    = (bf16_t*)(ws + take(M * 192 * 2));
  bf16_t* qkvB   = (bf16_t*)(ws + take(M * 576 * 2));
  bf16_t* obf    = (bf16_t*)(ws + take(M * 192 * 2));
  float*  outF   = (float* )(ws + take(M * 192 * 4));
  bf16_t* h2bf   = (bf16_t*)(ws + take(M * 192 * 2));
  bf16_t* gbf    = (bf16_t*)(ws + take(M * 384 * 2));
  bf16_t* finbf  = (bf16_t*)(ws + take(M * 192 * 2));
  float*  yF     = (float* )(ws + take(M * 256 * 4));
  bf16_t* B1p    = (bf16_t*)(ws + take((size_t)8  * 12 * 512 * 2));  // conv1^T: K=256,N=192
  bf16_t* qkvp   = (bf16_t*)(ws + take((size_t)6  * 36 * 512 * 2));  // K=192,N=576
  bf16_t* projp  = (bf16_t*)(ws + take((size_t)6  * 12 * 512 * 2));  // K=192,N=192
  bf16_t* ffn1p  = (bf16_t*)(ws + take((size_t)6  * 24 * 512 * 2));  // K=192,N=384
  bf16_t* ffn2p  = (bf16_t*)(ws + take((size_t)12 * 12 * 512 * 2));  // K=384,N=192
  bf16_t* conv2p = (bf16_t*)(ws + take((size_t)6  * 16 * 512 * 2));  // conv2^T: K=192,N=256

  // stage 0: pooling + weight packing (independent)
  pool_kernel<<<8192, 256, 0, stream>>>(x, xm);
  pack_b_kernel<<<8 * 12,  32, 0, stream>>>(conv1_w, B1p,    256, 192, 1, 12);
  pack_b_kernel<<<6 * 36,  32, 0, stream>>>(qkv_w,   qkvp,   192, 576, 0, 36);
  pack_b_kernel<<<6 * 12,  32, 0, stream>>>(proj_w,  projp,  192, 192, 0, 12);
  pack_b_kernel<<<6 * 24,  32, 0, stream>>>(ffn_w1,  ffn1p,  192, 384, 0, 24);
  pack_b_kernel<<<12 * 12, 32, 0, stream>>>(ffn_w2,  ffn2p,  384, 192, 0, 12);
  pack_b_kernel<<<6 * 16,  32, 0, stream>>>(conv2_w, conv2p, 192, 256, 1, 16);

  // tokens = pool(x) @ conv1^T ; LN1 -> tbf
  gemm_ln_kernel<<<512, 128, 0, stream>>>(xm, 256, B1p, 8, nullptr, nullptr, nullptr,
                                          tokens, ln1_g, ln1_b, tbf);
  // qkv = tbf @ qkv_w + qkv_b  (bf16 out)
  gemm_bias_bf16_kernel<<<dim3(512, 9), 128, 0, stream>>>(tbf, 192, qkvp, 6, 36, qkv_b, qkvB, 576);
  // attention -> obf
  attn_kernel<<<512, 128, 0, stream>>>(qkvB, rpb, obf);
  // out = tokens + alpha*(obf @ proj_w + proj_b) ; LN2 -> h2bf
  gemm_ln_kernel<<<512, 128, 0, stream>>>(obf, 192, projp, 6, proj_b, tokens, alpha,
                                          outF, ln2_g, ln2_b, h2bf);
  // g = gelu(h2bf @ ffn_w1 + b1)
  gemm_gelu_bf16_kernel<<<dim3(512, 6), 128, 0, stream>>>(h2bf, 192, ffn1p, 6, 24, ffn_b1, gbf, 384);
  // fin = out + g @ ffn_w2 + b2
  gemm_resid_bf16_kernel<<<dim3(512, 3), 128, 0, stream>>>(gbf, 384, ffn2p, 12, 12, ffn_b2, outF, finbf);
  // y = fin @ conv2^T
  gemm_bias_f32_kernel<<<dim3(512, 4), 128, 0, stream>>>(finbf, 192, conv2p, 6, 16, nullptr, yF, 256);
  // out = x + broadcast(y)
  final_add_kernel<<<32768, 256, 0, stream>>>(x, yF, outp);
}
